// GraphFrontDoor_38508676776172
// MI455X (gfx1250) — matrix operand order repaired and checked by reading, not compiled, not run on hardware
//
#include <hip/hip_runtime.h>

typedef __attribute__((ext_vector_type(16))) __bf16 v16bf;
typedef __attribute__((ext_vector_type(8)))  float  v8f;

// ---------- helpers ----------
__device__ __forceinline__ __bf16 f2bf(float f) {
  union { float f; unsigned u; } c; c.f = f;
  unsigned r = c.u + 0x7FFFu + ((c.u >> 16) & 1u);
  union { unsigned short s; __bf16 b; } o; o.s = (unsigned short)(r >> 16);
  return o.b;
}

__device__ __forceinline__ float wave_sum(float v) {
#pragma unroll
  for (int m = 16; m > 0; m >>= 1) v += __shfl_xor(v, m, 32);
  return v;
}

// ---------- degree / normalization ----------
__global__ void deg_kernel(const int* __restrict__ col, float* __restrict__ deg, int E) {
  int e = blockIdx.x * blockDim.x + threadIdx.x;
  if (e < E) atomicAdd(&deg[col[e]], 1.0f);
}

__global__ void dinv_kernel(float* __restrict__ deg, int N) {
  int i = blockIdx.x * blockDim.x + threadIdx.x;
  if (i < N) { float d = deg[i]; deg[i] = (d > 0.0f) ? rsqrtf(d) : 0.0f; }
}

__global__ void val_kernel(const int* __restrict__ row, const int* __restrict__ col,
                           const float* __restrict__ dinv, float* __restrict__ val, int E) {
  int e = blockIdx.x * blockDim.x + threadIdx.x;
  if (e < E) val[e] = dinv[col[e]] * dinv[row[e]];
}

// ---------- edge scatter: h_neigh[col] += val * h[row] ----------
__global__ void scatter_kernel(const int* __restrict__ row, const int* __restrict__ col,
                               const float* __restrict__ val, const float* __restrict__ h,
                               float* __restrict__ hn, int E) {
  long idx = (long)blockIdx.x * blockDim.x + threadIdx.x;
  long e = idx >> 4;
  if (e >= E) return;
  int d0 = (int)(idx & 15) * 4;
  const float4 hv = *(const float4*)(h + (size_t)row[e] * 64 + d0);
  float v = val[e];
  float* dst = hn + (size_t)col[e] * 64 + d0;
  atomicAdd(dst + 0, v * hv.x);
  atomicAdd(dst + 1, v * hv.y);
  atomicAdd(dst + 2, v * hv.z);
  atomicAdd(dst + 3, v * hv.w);
}

// ---------- pack f32 [K, Nout] weight into WMMA bf16 B-operand layout ----------
// out[(((t*nChunks)+c)*32 + lane)*16 + i] = W[c*32 + (lane>=16?16:0) + i][t*16 + lane%16]
__global__ void pack_weights(const float* __restrict__ W, int K, int Nout,
                             int nChunks, int nTiles, __bf16* __restrict__ out) {
  int idx = blockIdx.x * blockDim.x + threadIdx.x;
  int total = nChunks * nTiles * 32 * 16;
  if (idx >= total) return;
  int i = idx & 15;
  int lane = (idx >> 4) & 31;
  int rem = idx >> 9;
  int c = rem % nChunks;
  int t = rem / nChunks;
  int n = t * 16 + (lane & 15);
  int k = c * 32 + ((lane >> 4) & 1) * 16 + i;
  float v = (n < Nout && k < K) ? W[(size_t)k * Nout + n] : 0.0f;
  out[idx] = f2bf(v);
}

// ---------- unified WMMA GEMM: out = act(A @ W + bias + resid) ----------
// Geometry is compile-time so accumulators stay in fixed VGPRs (no movrel).
// A f32 row-major. If SPLIT, K chunks [0,NCHUNKS/2) read from A0, rest from A1 (ldA each).
// W pre-packed bf16 (B layout). One wave computes 16 rows x (NTILES*16) cols.
template<int NCHUNKS, int NTILES, bool SPLIT>
__global__ void gemm_wmma(const float* __restrict__ A0, const float* __restrict__ A1, int ldA,
                          const __bf16* __restrict__ Bp,
                          const float* __restrict__ bias, const float* __restrict__ resid, int ldRes,
                          float* __restrict__ out, int ldOut, int nOut, int doRelu, int Mrows) {
  int wave = (int)((blockIdx.x * blockDim.x + threadIdx.x) >> 5);
  int lane = threadIdx.x & 31;
  int mTiles = (Mrows + 15) >> 4;
  if (wave >= mTiles) return;
  int row0 = wave * 16;
  int m  = lane & 15;
  int hi = (lane >> 4) & 1;
  int arow = row0 + m; if (arow > Mrows - 1) arow = Mrows - 1;

  v8f acc[NTILES];
#pragma unroll
  for (int t = 0; t < NTILES; ++t) acc[t] = (v8f)(0.0f);

  const v16bf* Bv = (const v16bf*)Bp;

#pragma unroll
  for (int c = 0; c < NCHUNKS; ++c) {
    const float* Ap;
    int kof;
    if (SPLIT) {
      constexpr int HALF = NCHUNKS / 2;
      Ap  = (c < HALF) ? A0 : A1;
      kof = (c < HALF) ? c * 32 : (c - HALF) * 32;
    } else {
      Ap = A0; kof = c * 32;
    }
    const float* ar = Ap + (size_t)arow * ldA + kof + hi * 16;
    float4 x0 = *(const float4*)(ar);
    float4 x1 = *(const float4*)(ar + 4);
    float4 x2 = *(const float4*)(ar + 8);
    float4 x3 = *(const float4*)(ar + 12);
    v16bf a;
    a[0]  = f2bf(x0.x); a[1]  = f2bf(x0.y); a[2]  = f2bf(x0.z); a[3]  = f2bf(x0.w);
    a[4]  = f2bf(x1.x); a[5]  = f2bf(x1.y); a[6]  = f2bf(x1.z); a[7]  = f2bf(x1.w);
    a[8]  = f2bf(x2.x); a[9]  = f2bf(x2.y); a[10] = f2bf(x2.z); a[11] = f2bf(x2.w);
    a[12] = f2bf(x3.x); a[13] = f2bf(x3.y); a[14] = f2bf(x3.z); a[15] = f2bf(x3.w);

#pragma unroll
    for (int t = 0; t < NTILES; ++t) {
      v16bf b = Bv[(size_t)((t * NCHUNKS + c) * 32 + lane)];
      acc[t] = __builtin_amdgcn_wmma_f32_16x16x32_bf16(
          false, a, false, b, (short)0, acc[t], false, false);
    }
  }

#pragma unroll
  for (int t = 0; t < NTILES; ++t) {
    int col = t * 16 + m;
    if (col >= nOut) continue;
    float bv = bias ? bias[col] : 0.0f;
#pragma unroll
    for (int r = 0; r < 8; ++r) {
      int row = row0 + r + hi * 8;
      if (row >= Mrows) continue;
      float v = acc[t][r] + bv;
      if (resid) v += resid[(size_t)row * ldRes + col];
      if (doRelu) v = fmaxf(v, 0.0f);
      out[(size_t)row * ldOut + col] = v;
    }
  }
}

// ---------- LayerNorm over D=64, one wave per row (in-place OK) ----------
__global__ void ln_kernel(const float* __restrict__ in, const float* __restrict__ g,
                          const float* __restrict__ b, float* __restrict__ out, int M) {
  int wave = threadIdx.x >> 5;
  int lane = threadIdx.x & 31;
  int row = blockIdx.x * 8 + wave;
  if (row >= M) return;
  float2 x = *(const float2*)(in + (size_t)row * 64 + 2 * lane);
  float s  = wave_sum(x.x + x.y);
  float sq = wave_sum(x.x * x.x + x.y * x.y);
  float mean = s * (1.0f / 64.0f);
  float var  = sq * (1.0f / 64.0f) - mean * mean;
  float rs = rsqrtf(var + 1e-5f);
  float2 o;
  o.x = (x.x - mean) * rs * g[2 * lane]     + b[2 * lane];
  o.y = (x.y - mean) * rs * g[2 * lane + 1] + b[2 * lane + 1];
  *(float2*)(out + (size_t)row * 64 + 2 * lane) = o;
}

// ---------- k/v projection of the 8 contexts ----------
__global__ void kv_kernel(const float* __restrict__ ctx,
                          const float* __restrict__ kw, const float* __restrict__ kb,
                          const float* __restrict__ vw, const float* __restrict__ vb,
                          float* __restrict__ kbuf, float* __restrict__ vbuf) {
  int t = threadIdx.x;           // 512 threads
  int j = t >> 6, d = t & 63;
  float sk = kb[d], sv = vb[d];
  for (int i = 0; i < 64; ++i) {
    float cv = ctx[j * 64 + i];
    sk += cv * kw[i * 64 + d];
    sv += cv * vw[i * 64 + d];
  }
  kbuf[t] = sk; vbuf[t] = sv;
}

// ---------- attention over 8 contexts + blend + LN -> z_aug ----------
__global__ void attn_kernel(const float* __restrict__ q, const float* __restrict__ zc,
                            const float* __restrict__ kbuf, const float* __restrict__ vbuf,
                            const float* __restrict__ g, const float* __restrict__ b,
                            float* __restrict__ zaug, int M) {
  __shared__ float Ks[512], Vs[512];
  for (int i = threadIdx.x; i < 512; i += blockDim.x) { Ks[i] = kbuf[i]; Vs[i] = vbuf[i]; }
  __syncthreads();
  int wave = threadIdx.x >> 5;
  int lane = threadIdx.x & 31;
  int row = blockIdx.x * 8 + wave;
  if (row >= M) return;
  float2 q2 = *(const float2*)(q + (size_t)row * 64 + 2 * lane);
  float s[8];
#pragma unroll
  for (int j = 0; j < 8; ++j) {
    float p = q2.x * Ks[j * 64 + 2 * lane] + q2.y * Ks[j * 64 + 2 * lane + 1];
    s[j] = wave_sum(p) * 0.125f;   // 1/sqrt(64), GATE_TEMP=1
  }
  float mx = s[0];
#pragma unroll
  for (int j = 1; j < 8; ++j) mx = fmaxf(mx, s[j]);
  float den = 0.0f;
#pragma unroll
  for (int j = 0; j < 8; ++j) { s[j] = __expf(s[j] - mx); den += s[j]; }
  float inv = 1.0f / den;
  float cx = 0.0f, cy = 0.0f;
#pragma unroll
  for (int j = 0; j < 8; ++j) {
    float a = s[j] * inv;
    cx += a * Vs[j * 64 + 2 * lane];
    cy += a * Vs[j * 64 + 2 * lane + 1];
  }
  float2 z = *(const float2*)(zc + (size_t)row * 64 + 2 * lane);
  float px = z.x + 0.5f * cx;
  float py = z.y + 0.5f * cy;
  float sm  = wave_sum(px + py);
  float sq  = wave_sum(px * px + py * py);
  float mean = sm * (1.0f / 64.0f);
  float var  = sq * (1.0f / 64.0f) - mean * mean;
  float rs = rsqrtf(var + 1e-5f);
  float2 o;
  o.x = (px - mean) * rs * g[2 * lane]     + b[2 * lane];
  o.y = (py - mean) * rs * g[2 * lane + 1] + b[2 * lane + 1];
  *(float2*)(zaug + (size_t)row * 64 + 2 * lane) = o;
}

// ---------- host ----------
static inline size_t alignUp(size_t x, size_t a) { return (x + a - 1) & ~(a - 1); }

extern "C" void kernel_launch(void* const* d_in, const int* in_sizes, int n_in,
                              void* d_out, int out_size, void* d_ws, size_t ws_size,
                              hipStream_t stream) {
  const int N = in_sizes[0] / 128;
  const int E = in_sizes[1] / 2;

  const float* x        = (const float*)d_in[0];
  const int*   ei       = (const int*)  d_in[1];
  const float* contexts = (const float*)d_in[2];
  const float* W_in  = (const float*)d_in[3];
  const float* b_in  = (const float*)d_in[4];
  const float* gcn_w = (const float*)d_in[5];   // [2,128,64]
  const float* ca_w  = (const float*)d_in[6];
  const float* ca_b  = (const float*)d_in[7];
  const float* sa_w  = (const float*)d_in[8];
  const float* sa_b  = (const float*)d_in[9];
  const float* cn_g  = (const float*)d_in[10];
  const float* cn_b  = (const float*)d_in[11];
  const float* sn_g  = (const float*)d_in[12];
  const float* sn_b  = (const float*)d_in[13];
  const float* cls_w = (const float*)d_in[14];
  const float* cls_b = (const float*)d_in[15];
  const float* spu_w = (const float*)d_in[16];
  const float* spu_b = (const float*)d_in[17];
  const float* fd_w  = (const float*)d_in[18];
  const float* fd_b  = (const float*)d_in[19];
  const float* q_w   = (const float*)d_in[20];
  const float* q_b   = (const float*)d_in[21];
  const float* k_w   = (const float*)d_in[22];
  const float* k_b   = (const float*)d_in[23];
  const float* v_w   = (const float*)d_in[24];
  const float* v_b   = (const float*)d_in[25];
  const float* an_g  = (const float*)d_in[26];
  const float* an_b  = (const float*)d_in[27];

  const int* erow = ei;
  const int* ecol = ei + E;

  // workspace layout
  char* ws = (char*)d_ws;
  size_t off = 0;
  auto take = [&](size_t bytes) { size_t o = off; off = alignUp(off + bytes, 256); return (void*)(ws + o); };
  float* dinv = (float*)take((size_t)N * 4);
  float* val  = (float*)take((size_t)E * 4);
  float* buf0 = (float*)take((size_t)N * 64 * 4);  // h -> q
  float* buf1 = (float*)take((size_t)N * 64 * 4);  // h_neigh -> z_s -> z_aug
  float* buf2 = (float*)take((size_t)N * 64 * 4);  // z_c
  auto packSz = [](int K, int nTiles) { return (size_t)(K / 32) * nTiles * 32 * 16 * sizeof(__bf16); };
  __bf16* pWin  = (__bf16*)take(packSz(128, 4));
  __bf16* pG0   = (__bf16*)take(packSz(128, 4));
  __bf16* pG1   = (__bf16*)take(packSz(128, 4));
  __bf16* pCa   = (__bf16*)take(packSz(64, 4));
  __bf16* pSa   = (__bf16*)take(packSz(64, 4));
  __bf16* pQ    = (__bf16*)take(packSz(64, 4));
  __bf16* pCls  = (__bf16*)take(packSz(64, 3));
  __bf16* pSpu  = (__bf16*)take(packSz(64, 3));
  __bf16* pFd   = (__bf16*)take(packSz(64, 3));
  float* kbuf = (float*)take(512 * 4);
  float* vbuf = (float*)take(512 * 4);

  float* outCausal = (float*)d_out;
  float* outSpur   = outCausal + (size_t)N * 40;
  float* outFd     = outSpur   + (size_t)N * 40;

  const int TB = 256;
  int gE = (E + TB - 1) / TB;
  int gN = (N + TB - 1) / TB;
  int mTiles = (N + 15) / 16;
  int gGemm = (mTiles + 7) / 8;        // 8 waves / block
  int gRow  = (N + 7) / 8;             // 8 rows / block
  long scatTot = (long)E * 16;
  int gScat = (int)((scatTot + TB - 1) / TB);

  // degrees -> dinv -> edge values
  hipMemsetAsync(dinv, 0, (size_t)N * 4, stream);
  deg_kernel<<<gE, TB, 0, stream>>>(ecol, dinv, E);
  dinv_kernel<<<gN, TB, 0, stream>>>(dinv, N);
  val_kernel<<<gE, TB, 0, stream>>>(erow, ecol, dinv, val, E);

  // pack weights to WMMA B layout (bf16)
  auto pack = [&](const float* W, int K, int Nout, int nTiles, __bf16* dst) {
    int total = (K / 32) * nTiles * 512;
    pack_weights<<<(total + TB - 1) / TB, TB, 0, stream>>>(W, K, Nout, K / 32, nTiles, dst);
  };
  pack(W_in, 128, 64, 4, pWin);
  pack(gcn_w,            128, 64, 4, pG0);
  pack(gcn_w + 128 * 64, 128, 64, 4, pG1);
  pack(ca_w, 64, 64, 4, pCa);
  pack(sa_w, 64, 64, 4, pSa);
  pack(q_w,  64, 64, 4, pQ);
  pack(cls_w, 64, 40, 3, pCls);
  pack(spu_w, 64, 40, 3, pSpu);
  pack(fd_w,  64, 40, 3, pFd);

  // h = relu(x @ W_in + b_in)
  gemm_wmma<4, 4, false><<<gGemm, TB, 0, stream>>>(x, nullptr, 128, pWin,
                                                   b_in, nullptr, 64, buf0, 64, 64, 1, N);

  // 2 GCN layers: h = relu(concat(h_neigh, h) @ gcn_w[l] + h)   (in-place safe: row-private)
  const __bf16* pg[2] = { pG0, pG1 };
  for (int l = 0; l < 2; ++l) {
    hipMemsetAsync(buf1, 0, (size_t)N * 64 * 4, stream);
    scatter_kernel<<<gScat, TB, 0, stream>>>(erow, ecol, val, buf0, buf1, E);
    gemm_wmma<4, 4, true><<<gGemm, TB, 0, stream>>>(buf1, buf0, 64, pg[l],
                                                    nullptr, buf0, 64, buf0, 64, 64, 1, N);
  }

  // z_c = LN(h + h@ca_w + ca_b), z_s = LN(h + h@sa_w + sa_b)
  gemm_wmma<2, 4, false><<<gGemm, TB, 0, stream>>>(buf0, nullptr, 64, pCa,
                                                   ca_b, buf0, 64, buf2, 64, 64, 0, N);
  ln_kernel<<<gRow, TB, 0, stream>>>(buf2, cn_g, cn_b, buf2, N);
  gemm_wmma<2, 4, false><<<gGemm, TB, 0, stream>>>(buf0, nullptr, 64, pSa,
                                                   sa_b, buf0, 64, buf1, 64, 64, 0, N);
  ln_kernel<<<gRow, TB, 0, stream>>>(buf1, sn_g, sn_b, buf1, N);

  // logits heads
  gemm_wmma<2, 3, false><<<gGemm, TB, 0, stream>>>(buf2, nullptr, 64, pCls,
                                                   cls_b, nullptr, 64, outCausal, 40, 40, 0, N);
  gemm_wmma<2, 3, false><<<gGemm, TB, 0, stream>>>(buf1, nullptr, 64, pSpu,
                                                   spu_b, nullptr, 64, outSpur, 40, 40, 0, N);

  // q = z_c @ q_w + q_b  (h in buf0 is dead now)
  gemm_wmma<2, 4, false><<<gGemm, TB, 0, stream>>>(buf2, nullptr, 64, pQ,
                                                   q_b, nullptr, 64, buf0, 64, 64, 0, N);

  // contexts k/v, attention + blend + LN -> z_aug (into buf1; z_s dead)
  kv_kernel<<<1, 512, 0, stream>>>(contexts, k_w, k_b, v_w, v_b, kbuf, vbuf);
  attn_kernel<<<gRow, TB, 0, stream>>>(buf0, buf2, kbuf, vbuf, an_g, an_b, buf1, N);

  // fd_logits = z_aug @ fd_w + fd_b
  gemm_wmma<2, 3, false><<<gGemm, TB, 0, stream>>>(buf1, nullptr, 64, pFd,
                                                   fd_b, nullptr, 64, outFd, 40, 40, 0, N);

  (void)n_in; (void)out_size; (void)ws_size; (void)contexts;
}
